// CustomGPT2Attention_46703474377385
// MI455X (gfx1250) — compile-verified, matrix operand order checked by reading, not run on hardware
//
#include <hip/hip_runtime.h>

// ---------------------------------------------------------------------------
// GPT-2 attention block for MI455X (gfx1250), bf16 WMMA path.
// B=4, S=2048, D=1024, H=16, HD=64.
// Round 2: register-blocked GEMMs (4x2 frags/wave), async-to-LDS double-
// buffered K/V staging in the flash-attention kernel, V stored pre-transposed.
// ---------------------------------------------------------------------------

typedef __attribute__((ext_vector_type(16))) __bf16 v16bf;
typedef __attribute__((ext_vector_type(8)))  float  v8f;

#define S_LEN  2048
#define DMODEL 1024
#define NHEAD  16
#define HDIM   64
#define BATCH  4
#define MTOT   (BATCH * S_LEN)   // 8192 rows

// fp32 -> bf16 with round-to-nearest-even
__device__ __forceinline__ unsigned short f32_bf16(float f) {
    union { float f; unsigned u; } v; v.f = f;
    unsigned r = v.u + 0x7fffu + ((v.u >> 16) & 1u);
    return (unsigned short)(r >> 16);
}

// Load one 16x32 bf16 WMMA fragment (A-layout; identical indexing serves as
// B-layout when the source matrix is stored transposed, i.e. B[k][n]=M[n][k]).
// Per ISA 7.12.2: lane L -> row = base + (L&15); K-offset = (L>>4)*8;
// VGPRs0-3 = 8 contiguous K elems (16B), VGPRs4-7 = same at K+16 (+32B).
__device__ __forceinline__ v16bf load_frag(const unsigned short* base, int ld,
                                           int row, int k0) {
    int lane = threadIdx.x & 31;
    const unsigned short* p =
        base + (size_t)(row + (lane & 15)) * ld + k0 + ((lane >> 4) << 3);
    union { struct { uint4 a, b; } u; v16bf v; } r;
    r.u.a = *reinterpret_cast<const uint4*>(p);
    r.u.b = *reinterpret_cast<const uint4*>(p + 16);
    return r.v;
}

__device__ __forceinline__ v8f wmma_bf16(v16bf a, v16bf b, v8f c) {
    return __builtin_amdgcn_wmma_f32_16x16x32_bf16(
        false, a, false, b, (short)0, c, false, false);
}

// Async copy 16B global -> LDS (ASYNCcnt-tracked, no VGPR round-trip).
// lds_off = wave-relative LDS byte address (low 32 bits of generic pointer).
__device__ __forceinline__ void async_b128(unsigned short* lds_ptr,
                                           const unsigned short* gptr) {
    unsigned lds_off = (unsigned)(size_t)lds_ptr;
    asm volatile("global_load_async_to_lds_b128 %0, %1, off"
                 :: "v"(lds_off), "v"(gptr) : "memory");
}
__device__ __forceinline__ void wait_async0() {
    asm volatile("s_wait_asynccnt 0x0" ::: "memory");
}

// ---------------------------------------------------------------------------
// Kernel 1: fp32 -> bf16 elementwise convert (hidden_states)
// ---------------------------------------------------------------------------
__global__ void __launch_bounds__(256) cvt_x(const float* __restrict__ x,
                                             unsigned short* __restrict__ xb,
                                             int n) {
    int i = blockIdx.x * 256 + threadIdx.x;
    if (i < n) xb[i] = f32_bf16(x[i]);
}

// ---------------------------------------------------------------------------
// Kernel 2: transpose + convert the 4 weight matrices: WT[n][k] = W[k][n]
// ---------------------------------------------------------------------------
__global__ void __launch_bounds__(256) cvt_wT(
    const float* __restrict__ w0, const float* __restrict__ w1,
    const float* __restrict__ w2, const float* __restrict__ w3,
    unsigned short* __restrict__ o0, unsigned short* __restrict__ o1,
    unsigned short* __restrict__ o2, unsigned short* __restrict__ o3) {
    const float* src; unsigned short* dst;
    switch (blockIdx.y) {
        case 0: src = w0; dst = o0; break;
        case 1: src = w1; dst = o1; break;
        case 2: src = w2; dst = o2; break;
        default: src = w3; dst = o3; break;
    }
    int i = blockIdx.x * 256 + threadIdx.x;       // 0 .. 1M-1
    int n = i >> 10, k = i & 1023;
    dst[i] = f32_bf16(src[k * DMODEL + n]);
}

// ---------------------------------------------------------------------------
// Kernel 3: fused QKV projection GEMM, register-blocked.
// Wave tile = 64 rows x 32 cols (4 A-frags x 2 B-frags -> 8 WMMA / k-step).
// Block = 8 waves as 2(M) x 4(N) -> 128x128 tile. Q,K stored [B,H,S,HD];
// V stored transposed [B,H,HD,S] so attention can async-stage V^T directly.
// ---------------------------------------------------------------------------
__global__ void __launch_bounds__(256) qkv_gemm(
    const unsigned short* __restrict__ xb,
    const unsigned short* __restrict__ wqT, const unsigned short* __restrict__ wkT,
    const unsigned short* __restrict__ wvT,
    const float* __restrict__ bq, const float* __restrict__ bk,
    const float* __restrict__ bv,
    unsigned short* __restrict__ qd, unsigned short* __restrict__ kd,
    unsigned short* __restrict__ vd) {
    const unsigned short* wT; const float* bias; unsigned short* dst;
    if (blockIdx.z == 0)      { wT = wqT; bias = bq; dst = qd; }
    else if (blockIdx.z == 1) { wT = wkT; bias = bk; dst = kd; }
    else                      { wT = wvT; bias = bv; dst = vd; }

    int wave = threadIdx.x >> 5;
    int lane = threadIdx.x & 31;
    int m0 = blockIdx.x * 128 + (wave >> 2) * 64;   // 4 row tiles
    int n0 = blockIdx.y * 128 + (wave & 3) * 32;    // 2 col tiles

    v8f acc[4][2] = {};
    for (int k0 = 0; k0 < DMODEL; k0 += 32) {
        v16bf a[4], bfr[2];
#pragma unroll
        for (int i = 0; i < 4; ++i) a[i] = load_frag(xb, DMODEL, m0 + 16 * i, k0);
#pragma unroll
        for (int j = 0; j < 2; ++j) bfr[j] = load_frag(wT, DMODEL, n0 + 16 * j, k0);
#pragma unroll
        for (int i = 0; i < 4; ++i)
#pragma unroll
            for (int j = 0; j < 2; ++j)
                acc[i][j] = wmma_bf16(a[i], bfr[j], acc[i][j]);
    }

#pragma unroll
    for (int j = 0; j < 2; ++j) {
        int jc = n0 + 16 * j + (lane & 15);   // global output column
        float bj = bias[jc];
        int h = jc >> 6, hd = jc & 63;
#pragma unroll
        for (int i = 0; i < 4; ++i)
#pragma unroll
            for (int r = 0; r < 8; ++r) {
                int sg = m0 + 16 * i + r + ((lane >> 4) << 3);
                int b_ = sg >> 11, s = sg & 2047;
                unsigned short val = f32_bf16(acc[i][j][r] + bj);
                if (blockIdx.z == 2)  // V: transposed [bh][hd][s]
                    dst[(((size_t)(b_ * NHEAD + h)) * HDIM + hd) * S_LEN + s] = val;
                else                  // Q,K: [bh][s][hd]
                    dst[(((size_t)(b_ * NHEAD + h)) * S_LEN + s) * HDIM + hd] = val;
            }
    }
}

// ---------------------------------------------------------------------------
// Kernel 4: causal flash attention, one block = 128 query rows of one (b,h).
// 8 waves x 16 query rows. Key loop step 32; K tile [32][64] and V^T tile
// [64][32] staged via double-buffered global_load_async_to_lds_b128.
// ---------------------------------------------------------------------------
__global__ void __launch_bounds__(256) attn(
    const unsigned short* __restrict__ qb,
    const unsigned short* __restrict__ kbuf,
    const unsigned short* __restrict__ vtbuf,
    unsigned short* __restrict__ ab) {
    __shared__ __align__(16) unsigned short k_lds[2][32 * 64];   // [key][hd]
    __shared__ __align__(16) unsigned short vt_lds[2][64 * 32];  // [hd][key]
    __shared__ __align__(16) unsigned short p_lds[8 * 16 * 32];  // per-wave P

    int wave = threadIdx.x >> 5;
    int lane = threadIdx.x & 31;
    int bh = blockIdx.y;
    int b = bh >> 4, h = bh & 15;
    const unsigned short* Q  = qb    + (size_t)bh * S_LEN * HDIM;
    const unsigned short* K  = kbuf  + (size_t)bh * S_LEN * HDIM;
    const unsigned short* Vt = vtbuf + (size_t)bh * HDIM * S_LEN;
    int qblock = blockIdx.x * 128;
    int qbase  = qblock + wave * 16;
    int qlast  = qbase + 15;

    v16bf aq0 = load_frag(Q, HDIM, qbase, 0);
    v16bf aq1 = load_frag(Q, HDIM, qbase, 32);

    v8f acc0 = {}, acc1 = {}, acc2 = {}, acc3 = {};
    float mrow[8], lrow[8];
    for (int r = 0; r < 8; ++r) { mrow[r] = -1e30f; lrow[r] = 0.f; }

    unsigned short* pw = p_lds + wave * (16 * 32);
    int kend = qblock + 128;                   // causal upper bound for block
    int t = threadIdx.x;
    int krow = t >> 3, kcol = (t & 7) * 8;     // K tile: 32x64, 16B per thread
    int vrow = t >> 2, vcol = (t & 3) * 8;     // V^T tile: 64x32, 16B per thread

    // prologue: stage tile 0 into buffer 0
    async_b128(&k_lds[0][krow * 64 + kcol], K + (size_t)krow * HDIM + kcol);
    async_b128(&vt_lds[0][vrow * 32 + vcol], Vt + (size_t)vrow * S_LEN + vcol);

    int nbuf = 0;
    for (int kt = 0; kt < kend; kt += 32) {
        wait_async0();          // this wave's async copies have landed in LDS
        __syncthreads();        // all waves' copies visible; prev reads done
        if (kt + 32 < kend) {   // stage next tile into the other buffer
            async_b128(&k_lds[nbuf ^ 1][krow * 64 + kcol],
                       K + (size_t)(kt + 32 + krow) * HDIM + kcol);
            async_b128(&vt_lds[nbuf ^ 1][vrow * 32 + vcol],
                       Vt + (size_t)vrow * S_LEN + (kt + 32) + vcol);
        }
        if (kt <= qlast) {
            const unsigned short* kt_l = k_lds[nbuf];
            const unsigned short* vt_l = vt_lds[nbuf];
            // scores: 16x32 tile = two 16x16 C frags, K=64 via chained WMMA
            v8f s0 = {}, s1 = {};
            s0 = wmma_bf16(aq0, load_frag(kt_l, 64, 0, 0),   s0);
            s0 = wmma_bf16(aq1, load_frag(kt_l, 64, 0, 32),  s0);
            s1 = wmma_bf16(aq0, load_frag(kt_l, 64, 16, 0),  s1);
            s1 = wmma_bf16(aq1, load_frag(kt_l, 64, 16, 32), s1);

            int j0 = kt + (lane & 15);
#pragma unroll
            for (int r = 0; r < 8; ++r) {
                int row = qbase + r + ((lane >> 4) << 3);
                float x0 = s0[r] * 0.125f;     // 1/sqrt(64)
                float x1 = s1[r] * 0.125f;
                if (j0 > row)      x0 = -10000.f;
                if (j0 + 16 > row) x1 = -10000.f;
                float tm = fmaxf(x0, x1);
#pragma unroll
                for (int m = 1; m < 16; m <<= 1)
                    tm = fmaxf(tm, __shfl_xor(tm, m, 32));
                float mn = fmaxf(mrow[r], tm);
                float al = __expf(mrow[r] - mn);
                float p0 = __expf(x0 - mn);
                float p1 = __expf(x1 - mn);
                float ps = p0 + p1;
#pragma unroll
                for (int m = 1; m < 16; m <<= 1)
                    ps += __shfl_xor(ps, m, 32);
                lrow[r] = lrow[r] * al + ps;
                mrow[r] = mn;
                acc0[r] *= al; acc1[r] *= al; acc2[r] *= al; acc3[r] *= al;
                // bounce P (C-layout) through wave-private LDS to A-layout
                int pr = r + ((lane >> 4) << 3);
                pw[pr * 32 + (lane & 15)]      = f32_bf16(p0);
                pw[pr * 32 + 16 + (lane & 15)] = f32_bf16(p1);
            }
            asm volatile("s_wait_dscnt 0x0" ::: "memory");  // wave-private RAW
            v16bf pa = load_frag(pw, 32, 0, 0);
            acc0 = wmma_bf16(pa, load_frag(vt_l, 32, 0,  0), acc0);
            acc1 = wmma_bf16(pa, load_frag(vt_l, 32, 16, 0), acc1);
            acc2 = wmma_bf16(pa, load_frag(vt_l, 32, 32, 0), acc2);
            acc3 = wmma_bf16(pa, load_frag(vt_l, 32, 48, 0), acc3);
        }
        nbuf ^= 1;
    }

    // normalize and store merged-head bf16: ab[b][s][h*64 + n]
#pragma unroll
    for (int r = 0; r < 8; ++r) {
        float inv = 1.f / lrow[r];
        int sg = qbase + r + ((lane >> 4) << 3);
        size_t base = ((size_t)b * S_LEN + sg) * DMODEL + h * HDIM + (lane & 15);
        ab[base]      = f32_bf16(acc0[r] * inv);
        ab[base + 16] = f32_bf16(acc1[r] * inv);
        ab[base + 32] = f32_bf16(acc2[r] * inv);
        ab[base + 48] = f32_bf16(acc3[r] * inv);
    }
}

// ---------------------------------------------------------------------------
// Kernel 5: output projection, register-blocked, fp32 result: A @ Wo + bo
// ---------------------------------------------------------------------------
__global__ void __launch_bounds__(256) oproj(
    const unsigned short* __restrict__ ab,
    const unsigned short* __restrict__ woT,
    const float* __restrict__ bo,
    float* __restrict__ out) {
    int wave = threadIdx.x >> 5;
    int lane = threadIdx.x & 31;
    int m0 = blockIdx.x * 128 + (wave >> 2) * 64;
    int n0 = blockIdx.y * 128 + (wave & 3) * 32;

    v8f acc[4][2] = {};
    for (int k0 = 0; k0 < DMODEL; k0 += 32) {
        v16bf a[4], bfr[2];
#pragma unroll
        for (int i = 0; i < 4; ++i) a[i] = load_frag(ab, DMODEL, m0 + 16 * i, k0);
#pragma unroll
        for (int j = 0; j < 2; ++j) bfr[j] = load_frag(woT, DMODEL, n0 + 16 * j, k0);
#pragma unroll
        for (int i = 0; i < 4; ++i)
#pragma unroll
            for (int j = 0; j < 2; ++j)
                acc[i][j] = wmma_bf16(a[i], bfr[j], acc[i][j]);
    }
#pragma unroll
    for (int j = 0; j < 2; ++j) {
        int jc = n0 + 16 * j + (lane & 15);
        float bj = bo[jc];
#pragma unroll
        for (int i = 0; i < 4; ++i)
#pragma unroll
            for (int r = 0; r < 8; ++r) {
                int sg = m0 + 16 * i + r + ((lane >> 4) << 3);
                out[(size_t)sg * DMODEL + jc] = acc[i][j][r] + bj;
            }
    }
}

// ---------------------------------------------------------------------------
// Workspace layout (bytes):
//   0         : Xbf  [8192*1024] bf16 (16 MB)  -- reused as Abf after QKV
//   16777216  : WqT (2 MB)   18874368 : WkT (2 MB)
//   20971520  : WvT (2 MB)   23068672 : WoT (2 MB)
//   25165824  : Qbf [64][2048][64] (16 MB)
//   41943040  : Kbf [64][2048][64] (16 MB)
//   58720256  : Vtbf [64][64][2048] (16 MB, transposed)
//   total     : 75,497,472 bytes
// ---------------------------------------------------------------------------
extern "C" void kernel_launch(void* const* d_in, const int* in_sizes, int n_in,
                              void* d_out, int out_size, void* d_ws, size_t ws_size,
                              hipStream_t stream) {
    const float* hs = (const float*)d_in[0];
    const float* Wq = (const float*)d_in[1];
    const float* bq = (const float*)d_in[2];
    const float* Wk = (const float*)d_in[3];
    const float* bk = (const float*)d_in[4];
    const float* Wv = (const float*)d_in[5];
    const float* bv = (const float*)d_in[6];
    const float* Wo = (const float*)d_in[7];
    const float* bo = (const float*)d_in[8];

    char* ws = (char*)d_ws;
    unsigned short* xb  = (unsigned short*)(ws);
    unsigned short* wqT = (unsigned short*)(ws + 16777216);
    unsigned short* wkT = (unsigned short*)(ws + 18874368);
    unsigned short* wvT = (unsigned short*)(ws + 20971520);
    unsigned short* woT = (unsigned short*)(ws + 23068672);
    unsigned short* qbp = (unsigned short*)(ws + 25165824);
    unsigned short* kbp = (unsigned short*)(ws + 41943040);
    unsigned short* vtp = (unsigned short*)(ws + 58720256);
    unsigned short* abp = xb;   // alias: X no longer needed after QKV

    cvt_x<<<dim3((MTOT * DMODEL) / 256), 256, 0, stream>>>(hs, xb, MTOT * DMODEL);
    cvt_wT<<<dim3((DMODEL * DMODEL) / 256, 4), 256, 0, stream>>>(
        Wq, Wk, Wv, Wo, wqT, wkT, wvT, woT);
    qkv_gemm<<<dim3(MTOT / 128, DMODEL / 128, 3), 256, 0, stream>>>(
        xb, wqT, wkT, wvT, bq, bk, bv, qbp, kbp, vtp);
    attn<<<dim3(S_LEN / 128, BATCH * NHEAD), 256, 0, stream>>>(qbp, kbp, vtp, abp);
    oproj<<<dim3(MTOT / 128, DMODEL / 128), 256, 0, stream>>>(
        abp, woT, bo, (float*)d_out);
}
